// long_short_71090298683719
// MI455X (gfx1250) — matrix operand order repaired and checked by reading, not compiled, not run on hardware
//
#include <hip/hip_runtime.h>
#include <hip/hip_bf16.h>

typedef __attribute__((ext_vector_type(16))) __bf16 v16bf;
typedef __attribute__((ext_vector_type(8)))  float  v8f;

__device__ __forceinline__ unsigned short f2bf(float f) {
    unsigned int u = __builtin_bit_cast(unsigned int, f);
    unsigned int r = u + 0x7FFFu + ((u >> 16) & 1u);
    return (unsigned short)(r >> 16);
}

__device__ __forceinline__ float sigf(float x) { return 1.0f / (1.0f + __expf(-x)); }

// ---------------------------------------------------------------------------
// 0) bulk f32 -> bf16 conversion (packs 4 per thread)
// ---------------------------------------------------------------------------
__global__ void cvt_bf16_kernel(const float* __restrict__ in, unsigned short* __restrict__ out,
                                long n4) {
    long i = (long)blockIdx.x * blockDim.x + threadIdx.x;
    if (i >= n4) return;
    float4 v = ((const float4*)in)[i];
    union { unsigned short s[4]; uint2 u; } pk;
    pk.s[0] = f2bf(v.x); pk.s[1] = f2bf(v.y); pk.s[2] = f2bf(v.z); pk.s[3] = f2bf(v.w);
    ((uint2*)out)[i] = pk.u;
}

// ---------------------------------------------------------------------------
// 1) Build LSTM inputs
// ---------------------------------------------------------------------------
__global__ void build_in_kernel(const int* __restrict__ tok, const int* __restrict__ user,
                                const int* __restrict__ timei,
                                const float* __restrict__ emb_tok, int Etok,
                                const float* __restrict__ emb_user_,
                                const float* __restrict__ emb_hour_,
                                float* __restrict__ out, int IN, int total, int T) {
    int idx = blockIdx.x * blockDim.x + threadIdx.x;
    if (idx >= total) return;
    int f = idx % IN;
    int bt = idx / IN;
    int b = bt / T;
    float v;
    if (f < Etok)            v = emb_tok[(long)tok[bt] * Etok + f];
    else if (f < Etok + 128) v = emb_user_[(long)user[b] * 128 + (f - Etok)];
    else                     v = emb_hour_[(long)timei[bt] * 32 + (f - Etok - 128)];
    out[idx] = v;
}

// ---------------------------------------------------------------------------
// 2) Two-layer LSTM, one block (256 threads) per batch element. H = 256.
// ---------------------------------------------------------------------------
__global__ __launch_bounds__(256) void lstm2_kernel(
    const float* __restrict__ x, int IN,
    const float* __restrict__ wih0, const float* __restrict__ whh0, const float* __restrict__ b0,
    const float* __restrict__ wih1, const float* __restrict__ whh1, const float* __restrict__ b1,
    float* __restrict__ out, int T) {
    const int H = 256;
    __shared__ float sx[320];
    __shared__ float h0[H], c0[H], h1[H], c1[H];
    int j = threadIdx.x;
    int b = blockIdx.x;
    h0[j] = 0.f; c0[j] = 0.f; h1[j] = 0.f; c1[j] = 0.f;
    __syncthreads();
    for (int t = 0; t < T; ++t) {
        for (int f = j; f < IN; f += 256) sx[f] = x[((long)(b * T + t)) * IN + f];
        __syncthreads();
        float gi = b0[j], gf = b0[j + 256], gg = b0[j + 512], go = b0[j + 768];
        for (int k = 0; k < IN; ++k) {
            float xv = sx[k];
            gi += wih0[(long)j * IN + k] * xv;
            gf += wih0[(long)(j + 256) * IN + k] * xv;
            gg += wih0[(long)(j + 512) * IN + k] * xv;
            go += wih0[(long)(j + 768) * IN + k] * xv;
        }
        for (int k = 0; k < H; ++k) {
            float hv = h0[k];
            gi += whh0[(long)j * H + k] * hv;
            gf += whh0[(long)(j + 256) * H + k] * hv;
            gg += whh0[(long)(j + 512) * H + k] * hv;
            go += whh0[(long)(j + 768) * H + k] * hv;
        }
        float cn0 = sigf(gf) * c0[j] + sigf(gi) * tanhf(gg);
        float hn0 = sigf(go) * tanhf(cn0);
        __syncthreads();
        c0[j] = cn0; h0[j] = hn0;
        __syncthreads();
        gi = b1[j]; gf = b1[j + 256]; gg = b1[j + 512]; go = b1[j + 768];
        for (int k = 0; k < H; ++k) {
            float xv = h0[k];
            gi += wih1[(long)j * H + k] * xv;
            gf += wih1[(long)(j + 256) * H + k] * xv;
            gg += wih1[(long)(j + 512) * H + k] * xv;
            go += wih1[(long)(j + 768) * H + k] * xv;
        }
        for (int k = 0; k < H; ++k) {
            float hv = h1[k];
            gi += whh1[(long)j * H + k] * hv;
            gf += whh1[(long)(j + 256) * H + k] * hv;
            gg += whh1[(long)(j + 512) * H + k] * hv;
            go += whh1[(long)(j + 768) * H + k] * hv;
        }
        float cn1 = sigf(gf) * c1[j] + sigf(gi) * tanhf(gg);
        float hn1 = sigf(go) * tanhf(cn1);
        __syncthreads();
        c1[j] = cn1; h1[j] = hn1;
        __syncthreads();
        out[((long)(b * T + t)) * H + j] = hn1;
    }
}

// ---------------------------------------------------------------------------
// 3a) attn_dist rows -> bf16 (feeds bf16-A GEMM)
// ---------------------------------------------------------------------------
__global__ void attn_dist_kernel(const int* __restrict__ inputs_poi, const int* __restrict__ v2c,
                                 const float* __restrict__ dmat,
                                 unsigned short* __restrict__ outA, int P) {
    int bt = blockIdx.x;
    int r = v2c[inputs_poi[bt]];
    const float* row = dmat + (long)r * P;
    for (int p = threadIdx.x; p < P; p += blockDim.x) {
        float v = row[p];
        v = (v == 0.0f) ? 9999999.99f : v;
        outA[(long)bt * P + p] = f2bf(1.0f / v);
    }
}

// ---------------------------------------------------------------------------
// 3b) gathered row softmax; bf16_out selects output type
// ---------------------------------------------------------------------------
__global__ __launch_bounds__(256) void softmax_rows_kernel(
    const int* __restrict__ tok, const int* __restrict__ v2c,
    const float* __restrict__ mat, int width, void* __restrict__ out, int bf16_out) {
    __shared__ float buf[8000];
    __shared__ float red[256];
    int bt = blockIdx.x;
    int tid = threadIdx.x;
    int r = tok[bt];
    if (v2c) r = v2c[r];
    const float* row = mat + (long)r * width;
    float lmax = -1e30f;
    for (int i = tid; i < width; i += 256) { float v = row[i]; buf[i] = v; lmax = fmaxf(lmax, v); }
    red[tid] = lmax;
    __syncthreads();
    for (int s = 128; s > 0; s >>= 1) {
        if (tid < s) red[tid] = fmaxf(red[tid], red[tid + s]);
        __syncthreads();
    }
    float mx = red[0];
    __syncthreads();
    float lsum = 0.f;
    for (int i = tid; i < width; i += 256) { float e = __expf(buf[i] - mx); buf[i] = e; lsum += e; }
    red[tid] = lsum;
    __syncthreads();
    for (int s = 128; s > 0; s >>= 1) {
        if (tid < s) red[tid] += red[tid + s];
        __syncthreads();
    }
    float inv = 1.0f / red[0];
    if (bf16_out) {
        unsigned short* o = (unsigned short*)out;
        for (int i = tid; i < width; i += 256) o[(long)bt * width + i] = f2bf(buf[i] * inv);
    } else {
        float* o = (float*)out;
        for (int i = tid; i < width; i += 256) o[(long)bt * width + i] = buf[i] * inv;
    }
}

// ---------------------------------------------------------------------------
// 4) long-term preference: hid -> alpha -> u_long
// ---------------------------------------------------------------------------
__global__ __launch_bounds__(256) void ulong_kernel(
    const int* __restrict__ inputs_user,
    const int* __restrict__ long_poi, const int* __restrict__ long_cat,
    const int* __restrict__ long_hour, const int* __restrict__ long_week,
    const float* __restrict__ emb_user_, const float* __restrict__ emb_poi_,
    const float* __restrict__ emb_cat_, const float* __restrict__ emb_hour_,
    const float* __restrict__ emb_week_,
    const float* __restrict__ Wp, const float* __restrict__ Wc,
    const float* __restrict__ Wt, const float* __restrict__ bias_u,
    float* __restrict__ u_long_out) {
    __shared__ float hid[64 * 128];
    __shared__ float uvec[128];
    __shared__ float alpha[64];
    __shared__ float score[64];
    int b = blockIdx.x, tid = threadIdx.x;
    int usr = inputs_user[b];
    if (tid < 128) uvec[tid] = emb_user_[(long)usr * 128 + tid];
    __syncthreads();
    for (int idx = tid; idx < 64 * 128; idx += 256) {
        int l = idx >> 7, u = idx & 127;
        int pi = long_poi[usr * 64 + l];
        int ci = long_cat[usr * 64 + l];
        int hi = long_hour[usr * 64 + l];
        int wi = long_week[usr * 64 + l];
        float s = bias_u[u];
        const float* lp = emb_poi_ + (long)pi * 128;
        for (int e = 0; e < 128; ++e) s += lp[e] * Wp[e * 128 + u];
        const float* lc = emb_cat_ + (long)ci * 64;
        for (int e = 0; e < 64; ++e) s += lc[e] * Wc[e * 128 + u];
        const float* lh = emb_hour_ + (long)hi * 32;
        for (int e = 0; e < 32; ++e) s += lh[e] * Wt[e * 128 + u];
        const float* lw = emb_week_ + (long)wi * 32;
        for (int e = 0; e < 32; ++e) s += lw[e] * Wt[(32 + e) * 128 + u];
        hid[idx] = fmaxf(s, 0.0f);
    }
    __syncthreads();
    if (tid < 64) {
        float s = 0.f;
        for (int u = 0; u < 128; ++u) s += hid[tid * 128 + u] * uvec[u];
        score[tid] = s;
    }
    __syncthreads();
    if (tid == 0) {
        float mx = -1e30f;
        for (int l = 0; l < 64; ++l) mx = fmaxf(mx, score[l]);
        float sm = 0.f;
        for (int l = 0; l < 64; ++l) { float e = __expf(score[l] - mx); alpha[l] = e; sm += e; }
        float inv = 1.0f / sm;
        for (int l = 0; l < 64; ++l) alpha[l] *= inv;
    }
    __syncthreads();
    if (tid < 256) {
        int d = tid;
        float s = 0.f;
        for (int l = 0; l < 64; ++l) {
            float feat;
            if (d < 128)      feat = emb_poi_[(long)long_poi[usr * 64 + l] * 128 + d];
            else if (d < 192) feat = emb_cat_[(long)long_cat[usr * 64 + l] * 64 + (d - 128)];
            else if (d < 224) feat = emb_hour_[(long)long_hour[usr * 64 + l] * 32 + (d - 192)];
            else              feat = emb_week_[(long)long_week[usr * 64 + l] * 32 + (d - 224)];
            s += alpha[l] * feat;
        }
        u_long_out[b * 256 + d] = s;
    }
}

// ---------------------------------------------------------------------------
// 5) WMMA bf16 GEMM:  C[M,N] = A[M,K] * W[N,K]^T + bias[N]
//    TBM x 128 x 32 tiles, 256 threads = 8 waves (2 x 4).
//    ABF/WBF select bf16 (ushort) vs f32 source for A / W.
// ---------------------------------------------------------------------------
#define BN 128
#define BK 32

template <int TBM, bool ABF, bool WBF>
__global__ __launch_bounds__(256) void gemm_wmma(
    const void* __restrict__ Av, const void* __restrict__ Wv,
    const float* __restrict__ bias, float* __restrict__ C,
    int M, int N, int K) {
    constexpr int MI = TBM / 32;
    __shared__ __align__(16) unsigned short As[TBM][BK + 8];  // [m][k] bf16, 80B row
    __shared__ __align__(16) unsigned short Bs[BK][BN + 8];   // [k][n] bf16

    int tid = threadIdx.x;
    int lane = tid & 31;
    int wid = tid >> 5;
    int wm = wid >> 2;
    int wn = wid & 3;
    int mblk = blockIdx.y * TBM;
    int nblk = blockIdx.x * BN;

    v8f acc[MI][2];
    #pragma unroll
    for (int mi = 0; mi < MI; ++mi)
        #pragma unroll
        for (int ni = 0; ni < 2; ++ni)
            acc[mi][ni] = (v8f){0.f, 0.f, 0.f, 0.f, 0.f, 0.f, 0.f, 0.f};

    int nk = (K + BK - 1) / BK;
    for (int kc = 0; kc < nk; ++kc) {
        int k0 = kc * BK;
        // ---- A tile ----
        if (ABF) {
            const unsigned short* A = (const unsigned short*)Av;
            for (int q = tid; q < TBM * 4; q += 256) {      // quads of 8 shorts
                int m = q >> 2;
                int kq = (q & 3) << 3;
                int gm = mblk + m, gk = k0 + kq;
                uint4 v; v.x = 0; v.y = 0; v.z = 0; v.w = 0;
                if (gm < M) {
                    if (gk + 7 < K) v = *(const uint4*)(A + (long)gm * K + gk);
                    else {
                        union { uint4 u; unsigned short s[8]; } t;
                        for (int i = 0; i < 8; ++i)
                            t.s[i] = (gk + i < K) ? A[(long)gm * K + gk + i] : 0;
                        v = t.u;
                    }
                }
                *(uint4*)&As[m][kq] = v;                    // ds_store_b128
            }
        } else {
            const float* A = (const float*)Av;
            for (int q = tid; q < TBM * 8; q += 256) {      // quads of 4 floats
                int m = q >> 3;
                int kq = (q & 7) << 2;
                int gm = mblk + m, gk = k0 + kq;
                float vx = 0.f, vy = 0.f, vz = 0.f, vw = 0.f;
                if (gm < M) {
                    if (gk + 3 < K) {
                        float4 v = *(const float4*)(A + (long)gm * K + gk);
                        vx = v.x; vy = v.y; vz = v.z; vw = v.w;
                    } else {
                        if (gk + 0 < K) vx = A[(long)gm * K + gk + 0];
                        if (gk + 1 < K) vy = A[(long)gm * K + gk + 1];
                        if (gk + 2 < K) vz = A[(long)gm * K + gk + 2];
                        if (gk + 3 < K) vw = A[(long)gm * K + gk + 3];
                    }
                }
                union { unsigned short s[4]; uint2 u; } pk;
                pk.s[0] = f2bf(vx); pk.s[1] = f2bf(vy); pk.s[2] = f2bf(vz); pk.s[3] = f2bf(vw);
                *(uint2*)&As[m][kq] = pk.u;
            }
        }
        // ---- W tile (transposed into Bs[k][n]) ----
        if (WBF) {
            const unsigned short* W = (const unsigned short*)Wv;
            for (int q = tid; q < BN * 4; q += 256) {       // quads of 8 shorts
                int n = q >> 2;
                int kq = (q & 3) << 3;
                int gn = nblk + n, gk = k0 + kq;
                union { uint4 u; unsigned short s[8]; } t;
                t.u.x = 0; t.u.y = 0; t.u.z = 0; t.u.w = 0;
                if (gn < N) {
                    if (gk + 7 < K) t.u = *(const uint4*)(W + (long)gn * K + gk);
                    else for (int i = 0; i < 8; ++i)
                        t.s[i] = (gk + i < K) ? W[(long)gn * K + gk + i] : 0;
                    if (gk + BK + 7 < K) __builtin_prefetch(W + (long)gn * K + gk + BK, 0, 1);
                }
                #pragma unroll
                for (int i = 0; i < 8; ++i) Bs[kq + i][n] = t.s[i];
            }
        } else {
            const float* W = (const float*)Wv;
            for (int q = tid; q < BN * 8; q += 256) {
                int n = q >> 3;
                int kq = (q & 7) << 2;
                int gn = nblk + n, gk = k0 + kq;
                float vx = 0.f, vy = 0.f, vz = 0.f, vw = 0.f;
                if (gn < N) {
                    if (gk + 3 < K) {
                        float4 v = *(const float4*)(W + (long)gn * K + gk);
                        vx = v.x; vy = v.y; vz = v.z; vw = v.w;
                    } else {
                        if (gk + 0 < K) vx = W[(long)gn * K + gk + 0];
                        if (gk + 1 < K) vy = W[(long)gn * K + gk + 1];
                        if (gk + 2 < K) vz = W[(long)gn * K + gk + 2];
                        if (gk + 3 < K) vw = W[(long)gn * K + gk + 3];
                    }
                    if (gk + BK + 3 < K) __builtin_prefetch(W + (long)gn * K + gk + BK, 0, 1);
                }
                Bs[kq + 0][n] = f2bf(vx); Bs[kq + 1][n] = f2bf(vy);
                Bs[kq + 2][n] = f2bf(vz); Bs[kq + 3][n] = f2bf(vw);
            }
        }
        __syncthreads();

        // ---- fragments + WMMAs ----
        union Frag { v16bf v; unsigned int w[8]; };
        Frag bfr[2];
        int bh = (lane >> 4) * 8;
        #pragma unroll
        for (int ni = 0; ni < 2; ++ni) {
            int nb = wn * 32 + ni * 16 + (lane & 15);
            #pragma unroll
            for (int i = 0; i < 8; ++i) {
                int kk = (i < 4) ? (bh + 2 * i) : (16 + bh + 2 * (i - 4));
                bfr[ni].w[i] = (unsigned int)Bs[kk][nb] | ((unsigned int)Bs[kk + 1][nb] << 16);
            }
        }
        #pragma unroll
        for (int mi = 0; mi < MI; ++mi) {
            int m = wm * (TBM / 2) + mi * 16 + (lane & 15);
            const unsigned int* arow = (const unsigned int*)&As[m][0];
            int lo = (lane >> 4) * 4;
            Frag af;
            #pragma unroll
            for (int i = 0; i < 4; ++i) {
                af.w[i]     = arow[lo + i];
                af.w[4 + i] = arow[8 + lo + i];
            }
            acc[mi][0] = __builtin_amdgcn_wmma_f32_16x16x32_bf16(
                false, af.v, false, bfr[0].v, (short)0, acc[mi][0], false, false);
            acc[mi][1] = __builtin_amdgcn_wmma_f32_16x16x32_bf16(
                false, af.v, false, bfr[1].v, (short)0, acc[mi][1], false, false);
        }
        __syncthreads();
    }

    // ---- store ----
    #pragma unroll
    for (int mi = 0; mi < MI; ++mi) {
        #pragma unroll
        for (int ni = 0; ni < 2; ++ni) {
            int n = nblk + wn * 32 + ni * 16 + (lane & 15);
            if (n >= N) continue;
            int mrow0 = mblk + wm * (TBM / 2) + mi * 16 + ((lane >> 4) ? 8 : 0);
            float bv = bias[n];
            #pragma unroll
            for (int r = 0; r < 8; ++r) {
                int m = mrow0 + r;
                if (m < M) C[(long)m * N + n] = acc[mi][ni][r] + bv;
            }
        }
    }
}

// ---------------------------------------------------------------------------
// 6) Combine: out1 weighting + three softmaxes over T=9, fused per (b, p).
// ---------------------------------------------------------------------------
__global__ void combine_kernel(const int* __restrict__ user,
                               const float* __restrict__ wpa, const float* __restrict__ wca,
                               const float* __restrict__ op, const float* __restrict__ oc,
                               const float* __restrict__ ol,
                               const float* __restrict__ dl, const float* __restrict__ fl,
                               const float* __restrict__ cl,
                               float* __restrict__ out, int P, int T) {
    long g = (long)blockIdx.x * blockDim.x + threadIdx.x;
    if (g >= 64L * P) return;
    int b = (int)(g / P);
    int p = (int)(g % P);
    int u = user[b];
    float wp = wpa[u], wc = wca[u];
    float base = (1.0f - wp - wc) * ol[(long)b * P + p];
    float o1[9], dv[9], fv[9], cv[9];
    float dmx = -1e30f, fmx = -1e30f, cmx = -1e30f;
    for (int t = 0; t < 9; ++t) {
        long idx = ((long)(b * T + t)) * P + p;
        o1[t] = op[idx] * wp + oc[idx] * wc + base;
        dv[t] = dl[idx]; fv[t] = fl[idx]; cv[t] = cl[idx];
        dmx = fmaxf(dmx, dv[t]); fmx = fmaxf(fmx, fv[t]); cmx = fmaxf(cmx, cv[t]);
    }
    float ds = 0.f, fs = 0.f, cs = 0.f;
    for (int t = 0; t < 9; ++t) {
        dv[t] = __expf(dv[t] - dmx); ds += dv[t];
        fv[t] = __expf(fv[t] - fmx); fs += fv[t];
        cv[t] = __expf(cv[t] - cmx); cs += cv[t];
    }
    float di = 1.0f / ds, fi = 1.0f / fs, ci = 1.0f / cs;
    for (int t = 0; t < 9; ++t) {
        long idx = ((long)(b * T + t)) * P + p;
        out[idx] = o1[t] + dv[t] * di + fv[t] * fi + cv[t] * ci;
    }
}

// ---------------------------------------------------------------------------
extern "C" void kernel_launch(void* const* d_in, const int* in_sizes, int n_in,
                              void* d_out, int out_size, void* d_ws, size_t ws_size,
                              hipStream_t stream) {
    const int B = 64, T = 9, P = 8000, Cc = 400, H = 256;
    const int IN_POI = 288, IN_CAT = 224;
    const int BT = B * T;  // 576

    const int*   inputs_poi  = (const int*)d_in[0];
    const int*   inputs_cat  = (const int*)d_in[1];
    const int*   inputs_wekn = (const int*)d_in[2];
    const int*   inputs_user = (const int*)d_in[3];
    const int*   inputs_time = (const int*)d_in[4];
    const int*   v2c         = (const int*)d_in[9];
    const float* dmat        = (const float*)d_in[10];
    const float* fmat        = (const float*)d_in[11];
    const float* ctmat       = (const float*)d_in[12];
    const int*   long_poi    = (const int*)d_in[13];
    const int*   long_cat    = (const int*)d_in[14];
    const int*   long_hour   = (const int*)d_in[15];
    const int*   long_week   = (const int*)d_in[16];
    const float* emb_user_   = (const float*)d_in[17];
    const float* emb_poi_    = (const float*)d_in[18];
    const float* emb_cat_    = (const float*)d_in[19];
    const float* emb_hour_   = (const float*)d_in[20];
    const float* emb_week_   = (const float*)d_in[21];
    const float* out_w_poi   = (const float*)d_in[23];
    const float* out_w_cat   = (const float*)d_in[24];
    const float* weight_poi  = (const float*)d_in[25];
    const float* weight_cat  = (const float*)d_in[26];
    const float* weight_time = (const float*)d_in[27];
    const float* bias_u      = (const float*)d_in[28];
    const float* pih0 = (const float*)d_in[29];
    const float* phh0 = (const float*)d_in[30];
    const float* pb0  = (const float*)d_in[31];
    const float* pih1 = (const float*)d_in[32];
    const float* phh1 = (const float*)d_in[33];
    const float* pb1  = (const float*)d_in[34];
    const float* cih0 = (const float*)d_in[35];
    const float* chh0 = (const float*)d_in[36];
    const float* cb0  = (const float*)d_in[37];
    const float* cih1 = (const float*)d_in[38];
    const float* chh1 = (const float*)d_in[39];
    const float* cb1  = (const float*)d_in[40];
    const float* fc_poi_w  = (const float*)d_in[41];
    const float* fc_poi_b  = (const float*)d_in[42];
    const float* fc_cat_w  = (const float*)d_in[43];
    const float* fc_cat_b  = (const float*)d_in[44];
    const float* fc_long_w = (const float*)d_in[45];
    const float* fc_long_b = (const float*)d_in[46];
    const float* fc_dis_w  = (const float*)d_in[47];
    const float* fc_dis_b  = (const float*)d_in[48];
    const float* fc_catf_w = (const float*)d_in[49];
    const float* fc_catf_b = (const float*)d_in[50];

    // ---- workspace layout (byte offsets, 256B aligned regions) ----
    char* base = (char*)d_ws;
    size_t off = 0;
    auto alloc = [&](size_t bytes) -> void* {
        void* p = base + off;
        off += (bytes + 255) & ~(size_t)255;
        return p;
    };
    float*          poi_in    = (float*)alloc((size_t)BT * IN_POI * 4);
    float*          cat_in    = (float*)alloc((size_t)BT * IN_CAT * 4);
    float*          hpoi      = (float*)alloc((size_t)BT * H * 4);
    float*          hcat      = (float*)alloc((size_t)BT * H * 4);
    // stacked bf16 A: attn_dist || attn_freq (must be one contiguous region)
    unsigned short* abf       = (unsigned short*)alloc((size_t)2 * BT * P * 2);
    float*          attn_catf = (float*)alloc((size_t)BT * Cc * 4);
    float*          u_long    = (float*)alloc((size_t)B * 256 * 4);
    float*          op_lin    = (float*)alloc((size_t)BT * P * 4);
    float*          oc_lin    = (float*)alloc((size_t)BT * P * 4);
    float*          ol_lin    = (float*)alloc((size_t)B * P * 4);
    // stacked output: d_lin || f_lin (one contiguous region)
    float*          d_lin     = (float*)alloc((size_t)2 * BT * P * 4);
    float*          f_lin     = d_lin + (size_t)BT * P;
    float*          c_lin     = (float*)alloc((size_t)BT * P * 4);
    unsigned short* adist_bf  = abf;
    unsigned short* afreq_bf  = abf + (size_t)BT * P;
    // optional 128MB bf16 image of fc_dis_w (use only if workspace allows)
    size_t wbf_bytes = (size_t)P * P * 2;
    bool use_wbf = (off + wbf_bytes) <= ws_size;
    unsigned short* wdis_bf = use_wbf ? (unsigned short*)alloc(wbf_bytes) : nullptr;

    // 0) one-time (per call) f32 -> bf16 conversion of the dominant weight matrix
    if (use_wbf) {
        long n4 = (long)P * P / 4;  // 16M float4 groups
        cvt_bf16_kernel<<<(unsigned)((n4 + 255) / 256), 256, 0, stream>>>(fc_dis_w, wdis_bf, n4);
    }
    // 1) LSTM input assembly
    {
        int tot = BT * IN_POI;
        build_in_kernel<<<(tot + 255) / 256, 256, 0, stream>>>(
            inputs_poi, inputs_user, inputs_time, emb_poi_, 128, emb_user_, emb_hour_,
            poi_in, IN_POI, tot, T);
        tot = BT * IN_CAT;
        build_in_kernel<<<(tot + 255) / 256, 256, 0, stream>>>(
            inputs_cat, inputs_user, inputs_time, emb_cat_, 64, emb_user_, emb_hour_,
            cat_in, IN_CAT, tot, T);
    }
    // 2) LSTMs
    lstm2_kernel<<<B, 256, 0, stream>>>(poi_in, IN_POI, pih0, phh0, pb0, pih1, phh1, pb1, hpoi, T);
    lstm2_kernel<<<B, 256, 0, stream>>>(cat_in, IN_CAT, cih0, chh0, cb0, cih1, chh1, cb1, hcat, T);
    // 3) attention rows (bf16 for the big GEMM, f32 for the small-K one)
    attn_dist_kernel<<<BT, 256, 0, stream>>>(inputs_poi, v2c, dmat, adist_bf, P);
    softmax_rows_kernel<<<BT, 256, 0, stream>>>(inputs_wekn, v2c, fmat, P, (void*)afreq_bf, 1);
    softmax_rows_kernel<<<BT, 256, 0, stream>>>(inputs_time, nullptr, ctmat, Cc, (void*)attn_catf, 0);
    // 4) long-term preference
    ulong_kernel<<<B, 256, 0, stream>>>(inputs_user, long_poi, long_cat, long_hour, long_week,
                                        emb_user_, emb_poi_, emb_cat_, emb_hour_, emb_week_,
                                        weight_poi, weight_cat, weight_time, bias_u, u_long);
    // 5) GEMMs via WMMA bf16
    {
        dim3 blk(256);
        dim3 gP((P + BN - 1) / BN, (BT + 127) / 128);        // 63 x 5  (BM=128)
        dim3 gB((P + BN - 1) / BN, 1);                       // 63 x 1
        dim3 gDF((P + BN - 1) / BN, (2 * BT + 255) / 256);   // 63 x 5  (BM=256, M=1152)
        gemm_wmma<128, false, false><<<gP, blk, 0, stream>>>(hpoi, fc_poi_w, fc_poi_b, op_lin, BT, P, H);
        gemm_wmma<128, false, false><<<gP, blk, 0, stream>>>(hcat, fc_cat_w, fc_cat_b, oc_lin, BT, P, H);
        gemm_wmma<128, false, false><<<gB, blk, 0, stream>>>(u_long, fc_long_w, fc_long_b, ol_lin, B, P, 256);
        // d_lin & f_lin share fc_dis_w: stacked M = 1152, bf16 A; bf16 W if it fits
        if (use_wbf) {
            gemm_wmma<256, true, true><<<gDF, blk, 0, stream>>>(abf, wdis_bf, fc_dis_b, d_lin, 2 * BT, P, P);
        } else {
            gemm_wmma<256, true, false><<<gDF, blk, 0, stream>>>(abf, fc_dis_w, fc_dis_b, d_lin, 2 * BT, P, P);
        }
        gemm_wmma<128, false, false><<<gP, blk, 0, stream>>>(attn_catf, fc_catf_w, fc_catf_b, c_lin, BT, P, Cc);
    }
    // 6) combine + per-(b,p) softmaxes over T
    {
        long tot = (long)B * P;
        combine_kernel<<<(unsigned)((tot + 255) / 256), 256, 0, stream>>>(
            inputs_user, out_w_poi, out_w_cat, op_lin, oc_lin, ol_lin,
            d_lin, f_lin, c_lin, (float*)d_out, P, T);
    }
}
